// ExportableGENConv_5377299054769
// MI455X (gfx1250) — compile-verified, matrix operand order checked by reading, not run on hardware
//
#include <hip/hip_runtime.h>
#include <math.h>

// ---------------------------------------------------------------------------
// GENConv fused pipeline for gfx1250 (MI455X).
// - All GEMMs via v_wmma_f32_16x16x32_bf16 (fp32 accumulate).
// - Weights pre-packed once per launch into bf16 WMMA B-layout (no per-wave
//   fp32->bf16 weight conversion, single contiguous 32B load per lane).
// - k2 neighbor-feature gather via GLOBAL_LOAD_ASYNC_TO_LDS_B128 (ASYNCcnt),
//   overlapped with the edge-attr WMMAs; s_wait_asynccnt before the softmax.
// ---------------------------------------------------------------------------

typedef __attribute__((ext_vector_type(16))) __bf16 v16bf;
typedef __attribute__((ext_vector_type(8)))  float  v8f;

#define NNODES 50000
#define KNBR   32
#define INC    128
#define H      64
#define H2     128
#define EDGED  32
#define NTILES (NNODES / 16)   // 3125 (exact)

// Packed-weight tile bases (each tile = 32 lanes * 16 bf16 = 512 elements)
#define PT_WSRC  0    // 4x4 tiles
#define PT_WDST  16   // 4x4
#define PT_WEDGE 32   // 4x1
#define PT_W1    36   // 8x2
#define PT_W2    52   // 4x4
#define PT_TOTAL 68

// ---- WMMA tile helpers (layouts per CDNA5 ISA §7.12.2) ---------

// A-tile 16x32 (M x K) from fp32 row-major [ld]; lanes 0-15 hold K {0-7,16-23}
// of row M=lane, lanes 16-31 hold K {8-15,24-31}.
__device__ __forceinline__ v16bf load_tileA(const float* base, int ld, int row0, int k0) {
  const int lane = threadIdx.x & 31;
  const int half = lane >> 4;
  const int m    = lane & 15;
  const float* p = base + (size_t)(row0 + m) * ld + k0 + 8 * half;
  v16bf a;
#pragma unroll
  for (int i = 0; i < 8; ++i) a[i]     = (__bf16)p[i];
#pragma unroll
  for (int i = 0; i < 8; ++i) a[8 + i] = (__bf16)p[16 + i];
  return a;
}

// B-tile 32x16 (K x N) of W^T, B(k,n)=W[n,k], W fp32 row-major [ldw].
__device__ __forceinline__ v16bf load_tileB(const float* W, int ldw, int n0, int k0) {
  const int lane = threadIdx.x & 31;
  const int half = lane >> 4;
  const int n    = lane & 15;
  const float* p = W + (size_t)(n0 + n) * ldw + k0 + 8 * half;
  v16bf b;
#pragma unroll
  for (int i = 0; i < 8; ++i) b[i]     = (__bf16)p[i];
#pragma unroll
  for (int i = 0; i < 8; ++i) b[8 + i] = (__bf16)p[16 + i];
  return b;
}

// Packed B-tile: one contiguous 32B v16bf per lane.
__device__ __forceinline__ v16bf load_packedB(const __bf16* __restrict__ P, int t) {
  const int lane = threadIdx.x & 31;
  return *(const v16bf*)(P + ((size_t)t * 32 + lane) * 16);
}

// D 16x16 f32: vgpr r holds M = r + 8*half, N = lane%16.
__device__ __forceinline__ void store_tileD(float* out, int ld, int row0, int n0, v8f c) {
  const int lane = threadIdx.x & 31;
  const int half = lane >> 4;
  const int n    = lane & 15;
#pragma unroll
  for (int r = 0; r < 8; ++r)
    out[(size_t)(row0 + r + 8 * half) * ld + n0 + n] = c[r];
}

// ---- K-pack: convert all weights to bf16 in WMMA B-layout -----------------
__global__ __launch_bounds__(128) void k_pack_weights(
    const float* __restrict__ w_src, const float* __restrict__ w_dst,
    const float* __restrict__ w_edge, const float* __restrict__ w1,
    const float* __restrict__ w2, __bf16* __restrict__ P) {
  const int t = blockIdx.x * 4 + (threadIdx.x >> 5);
  if (t >= PT_TOTAL) return;
  const float* W; int ldw, nt, kc;
  if (t < PT_WDST)       { int u = t - PT_WSRC;  W = w_src;  ldw = INC;   nt = u >> 2; kc = u & 3; }
  else if (t < PT_WEDGE) { int u = t - PT_WDST;  W = w_dst;  ldw = INC;   nt = u >> 2; kc = u & 3; }
  else if (t < PT_W1)    { int u = t - PT_WEDGE; W = w_edge; ldw = EDGED; nt = u;      kc = 0;     }
  else if (t < PT_W2)    { int u = t - PT_W1;    W = w1;     ldw = H;     nt = u >> 1; kc = u & 1; }
  else                   { int u = t - PT_W2;    W = w2;     ldw = H2;    nt = u >> 2; kc = u & 3; }
  v16bf b = load_tileB(W, ldw, nt * 16, kc * 32);
  *(v16bf*)(P + ((size_t)t * 32 + (threadIdx.x & 31)) * 16) = b;
}

// ---- K1: src = x@w_src^T, dst_feat = x@w_dst^T ----------------------------
__global__ __launch_bounds__(256) void k1_node_linear(
    const float* __restrict__ x, const __bf16* __restrict__ P,
    float* __restrict__ srcf, float* __restrict__ dstf) {
  const int tile = blockIdx.x * 8 + (threadIdx.x >> 5);
  if (tile >= NTILES) return;           // uniform per wave -> EXEC stays full
  const int row0 = tile * 16;

  v16bf A[4];
#pragma unroll
  for (int kc = 0; kc < 4; ++kc) A[kc] = load_tileA(x, INC, row0, kc * 32);

#pragma unroll
  for (int which = 0; which < 2; ++which) {
    const int base = which ? PT_WDST : PT_WSRC;
    float*    O    = which ? dstf    : srcf;
#pragma unroll
    for (int nt = 0; nt < 4; ++nt) {
      v8f c = {};
#pragma unroll
      for (int kc = 0; kc < 4; ++kc) {
        v16bf B = load_packedB(P, base + nt * 4 + kc);
        c = __builtin_amdgcn_wmma_f32_16x16x32_bf16(false, A[kc], false, B,
                                                    (short)0, c, false, false);
      }
      store_tileD(O, H, row0, nt * 16, c);
    }
  }
}

// ---- K0: zero BN accumulators --------------------------------------------
__global__ void k0_zero_stats(float* __restrict__ stats) {
  stats[threadIdx.x] = 0.0f;   // 256 threads: [0,128)=sum, [128,256)=sumsq
}

// ---- K2: fused edge matmul + async gather + masked softmax + aggregate ----
// One wave per node. Edges of node n are e = n*32+k, valid iff nbr[e] >= 0.
__global__ __launch_bounds__(128) void k2_edge(
    const float* __restrict__ edge_attr, const __bf16* __restrict__ P,
    const float* __restrict__ srcf, const float* __restrict__ dstf,
    const int* __restrict__ esrc, const int* __restrict__ nbr,
    float* __restrict__ outf) {
  __shared__ float xj[4][KNBR * H];   // gathered src features, 8KB/wave
  __shared__ int   sid[4][KNBR];
  __shared__ int   nb[4][KNBR];

  const int wave = threadIdx.x >> 5;
  const int lane = threadIdx.x & 31;
  const int half = lane >> 4;
  const int ncol = lane & 15;
  const int node = blockIdx.x * 4 + wave;
  const int e0   = node * KNBR;

  nb[wave][lane]  = nbr[e0 + lane];
  sid[wave][lane] = esrc[e0 + lane];
  __builtin_prefetch(edge_attr + (size_t)(e0 + 4 * KNBR) * EDGED, 0, 1);
  __syncthreads();

  // Async gather of the 32 neighbor feature rows (64 floats each) into LDS:
  // 16 x GLOBAL_LOAD_ASYNC_TO_LDS_B128, each moving 2 rows (lanes 0-15 row
  // 2m, lanes 16-31 row 2m+1, 16B per lane).  Tracked by ASYNCcnt; the
  // edge-attr WMMAs below execute while these are in flight.
  {
    const int rowh = lane >> 4;
    const int col4 = (lane & 15) * 4;
#pragma unroll
    for (int m2 = 0; m2 < 16; ++m2) {
      const int row = m2 * 2 + rowh;
      const float* g = srcf + (size_t)sid[wave][row] * H + col4;
      // Low 32 bits of a flat LDS pointer == wave-relative LDS offset (§10.2).
      const unsigned ldsoff =
          (unsigned)(unsigned long long)(const void*)&xj[wave][row * H + col4];
      asm volatile("global_load_async_to_lds_b128 %0, %1, off"
                   :: "v"(ldsoff), "v"((unsigned long long)g)
                   : "memory");
    }
  }

  v16bf B[4];
#pragma unroll
  for (int nt = 0; nt < 4; ++nt) B[nt] = load_packedB(P, PT_WEDGE + nt);

  v8f c[2][4];
#pragma unroll
  for (int mt = 0; mt < 2; ++mt) {
    v16bf A = load_tileA(edge_attr, EDGED, e0 + mt * 16, 0);
#pragma unroll
    for (int nt = 0; nt < 4; ++nt) {
      v8f z = {};
      c[mt][nt] = __builtin_amdgcn_wmma_f32_16x16x32_bf16(
          false, A, false, B[nt], (short)0, z, false, false);
    }
  }

  // Wait for the async gather, then build messages.
  asm volatile("s_wait_asynccnt 0x0" ::: "memory");

  float msg[2][4][8];
#pragma unroll
  for (int mt = 0; mt < 2; ++mt)
#pragma unroll
    for (int nt = 0; nt < 4; ++nt)
#pragma unroll
      for (int r = 0; r < 8; ++r) {
        const int m = mt * 16 + 8 * half + r;
        float v = c[mt][nt][r] + xj[wave][m * H + nt * 16 + ncol];
        msg[mt][nt][r] = fmaxf(v, 0.0f) + 1e-7f;   // relu + MSG_EPS
      }

  // Masked softmax over the 32 neighbor slots, per channel.  Column n lives
  // in lanes {lane, lane^16}: one shfl_xor(16) completes each reduction.
#pragma unroll
  for (int nt = 0; nt < 4; ++nt) {
    float mx = -3.402823466e38f;
#pragma unroll
    for (int mt = 0; mt < 2; ++mt)
#pragma unroll
      for (int r = 0; r < 8; ++r) {
        const int m = mt * 16 + 8 * half + r;
        mx = (nb[wave][m] >= 0) ? fmaxf(mx, msg[mt][nt][r]) : mx;
      }
    mx = fmaxf(mx, __shfl_xor(mx, 16));

    float s = 0.0f, ws = 0.0f;
#pragma unroll
    for (int mt = 0; mt < 2; ++mt)
#pragma unroll
      for (int r = 0; r < 8; ++r) {
        const int m = mt * 16 + 8 * half + r;
        const float a = (nb[wave][m] >= 0) ? __expf(msg[mt][nt][r] - mx) : 0.0f;
        s  += a;
        ws += msg[mt][nt][r] * a;
      }
    s  += __shfl_xor(s, 16);
    ws += __shfl_xor(ws, 16);

    if (half == 0) {
      const int col = nt * 16 + ncol;
      outf[(size_t)node * H + col] =
          ws / (s + 1e-16f) + dstf[(size_t)node * H + col];
    }
  }
}

// ---- K3: h = out@w1^T  +  per-channel sum / sumsq for BN ------------------
__global__ __launch_bounds__(256) void k3_mlp1(
    const float* __restrict__ outf, const __bf16* __restrict__ P,
    float* __restrict__ h, float* __restrict__ stats) {
  const int tile = blockIdx.x * 8 + (threadIdx.x >> 5);
  if (tile >= NTILES) return;
  const int row0 = tile * 16;
  const int lane = threadIdx.x & 31;
  const int half = lane >> 4;
  const int n    = lane & 15;

  v16bf A[2];
  A[0] = load_tileA(outf, H, row0, 0);
  A[1] = load_tileA(outf, H, row0, 32);

#pragma unroll
  for (int nt = 0; nt < 8; ++nt) {
    v8f c = {};
#pragma unroll
    for (int kc = 0; kc < 2; ++kc) {
      v16bf B = load_packedB(P, PT_W1 + nt * 2 + kc);
      c = __builtin_amdgcn_wmma_f32_16x16x32_bf16(false, A[kc], false, B,
                                                  (short)0, c, false, false);
    }
    store_tileD(h, H2, row0, nt * 16, c);

    float s1 = 0.0f, s2 = 0.0f;
#pragma unroll
    for (int r = 0; r < 8; ++r) { s1 += c[r]; s2 += c[r] * c[r]; }
    s1 += __shfl_xor(s1, 16);
    s2 += __shfl_xor(s2, 16);
    if (half == 0) {
      atomicAdd(&stats[nt * 16 + n], s1);
      atomicAdd(&stats[H2 + nt * 16 + n], s2);
    }
  }
}

// ---- K4: fold BN stats into per-channel scale/shift -----------------------
__global__ void k4_bn_finalize(const float* __restrict__ gamma,
                               const float* __restrict__ beta,
                               float* __restrict__ stats) {
  const int c = threadIdx.x;            // 128 channels
  const float invN = 1.0f / (float)NNODES;
  const float mean = stats[c] * invN;
  const float var  = stats[H2 + c] * invN - mean * mean;   // biased var
  const float sc   = gamma[c] * rsqrtf(var + 1e-5f);
  stats[256 + c] = sc;
  stats[384 + c] = beta[c] - mean * sc;
}

// A-tile load from h with BN scale/shift + ReLU fused into the bf16 convert.
__device__ __forceinline__ v16bf load_tileA_bn(const float* base, int row0, int k0,
                                               const float* __restrict__ scale,
                                               const float* __restrict__ shift) {
  const int lane = threadIdx.x & 31;
  const int half = lane >> 4;
  const int m    = lane & 15;
  const int kb   = k0 + 8 * half;
  const float* p = base + (size_t)(row0 + m) * H2 + kb;
  v16bf a;
#pragma unroll
  for (int i = 0; i < 8; ++i)
    a[i] = (__bf16)fmaxf(p[i] * scale[kb + i] + shift[kb + i], 0.0f);
#pragma unroll
  for (int i = 0; i < 8; ++i)
    a[8 + i] = (__bf16)fmaxf(p[16 + i] * scale[kb + 16 + i] + shift[kb + 16 + i], 0.0f);
  return a;
}

// ---- K5: y = relu(bn(h)) @ w2^T ------------------------------------------
__global__ __launch_bounds__(256) void k5_mlp2(
    const float* __restrict__ h, const __bf16* __restrict__ P,
    const float* __restrict__ stats, float* __restrict__ y) {
  const int tile = blockIdx.x * 8 + (threadIdx.x >> 5);
  if (tile >= NTILES) return;
  const int row0 = tile * 16;
  const float* scale = stats + 256;
  const float* shift = stats + 384;

  v16bf A[4];
#pragma unroll
  for (int kc = 0; kc < 4; ++kc)
    A[kc] = load_tileA_bn(h, row0, kc * 32, scale, shift);

#pragma unroll
  for (int nt = 0; nt < 4; ++nt) {
    v8f c = {};
#pragma unroll
    for (int kc = 0; kc < 4; ++kc) {
      v16bf B = load_packedB(P, PT_W2 + nt * 4 + kc);
      c = __builtin_amdgcn_wmma_f32_16x16x32_bf16(false, A[kc], false, B,
                                                  (short)0, c, false, false);
    }
    store_tileD(y, H, row0, nt * 16, c);
  }
}

// ---- launch ---------------------------------------------------------------
extern "C" void kernel_launch(void* const* d_in, const int* in_sizes, int n_in,
                              void* d_out, int out_size, void* d_ws, size_t ws_size,
                              hipStream_t stream) {
  (void)in_sizes; (void)n_in; (void)out_size; (void)ws_size;
  const float* x         = (const float*)d_in[0];
  const float* edge_attr = (const float*)d_in[1];
  const float* w_src     = (const float*)d_in[2];
  const float* w_dst     = (const float*)d_in[3];
  const float* w_edge    = (const float*)d_in[4];
  const float* w1        = (const float*)d_in[5];
  const float* gamma     = (const float*)d_in[6];
  const float* beta      = (const float*)d_in[7];
  const float* w2        = (const float*)d_in[8];
  const int*   eidx      = (const int*)d_in[9];    // [2,E]; row 0 = src
  const int*   nbr       = (const int*)d_in[10];   // [N,K]
  float* y = (float*)d_out;

  float* ws    = (float*)d_ws;
  float* srcf  = ws;                                // N*H
  float* dstf  = srcf + (size_t)NNODES * H;         // N*H
  float* outf  = dstf + (size_t)NNODES * H;         // N*H
  float* h     = outf + (size_t)NNODES * H;         // N*H2
  float* stats = h + (size_t)NNODES * H2;           // 512 floats
  __bf16* P    = (__bf16*)(stats + 512);            // 68*512 bf16 packed tiles

  const int g8 = (NTILES + 7) / 8;   // 391 blocks of 8 waves

  k_pack_weights<<<(PT_TOTAL + 3) / 4, 128, 0, stream>>>(w_src, w_dst, w_edge,
                                                         w1, w2, P);
  k1_node_linear<<<g8, 256, 0, stream>>>(x, P, srcf, dstf);
  k0_zero_stats<<<1, 256, 0, stream>>>(stats);
  k2_edge<<<NNODES / 4, 128, 0, stream>>>(edge_attr, P, srcf, dstf,
                                          eidx, nbr, outf);
  k3_mlp1<<<g8, 256, 0, stream>>>(outf, P, h, stats);
  k4_bn_finalize<<<1, 128, 0, stream>>>(gamma, beta, stats);
  k5_mlp2<<<g8, 256, 0, stream>>>(h, P, stats, y);
}
